// DUEHead_44444321579348
// MI455X (gfx1250) — matrix operand order repaired and checked by reading, not compiled, not run on hardware
//
#include <hip/hip_runtime.h>
#include <hip/hip_bf16.h>
#include <math.h>

// ---------------- types for WMMA ----------------
typedef __attribute__((ext_vector_type(16))) __bf16 bf16x16;
typedef __attribute__((ext_vector_type(8)))  __bf16 bf16x8;
typedef __attribute__((ext_vector_type(8)))  float  f32x8;

__device__ __forceinline__ float gelu_exact(float x) {
    return 0.5f * x * (1.0f + erff(x * 0.70710678118654752f));
}

// 32-byte global -> LDS async copy (2x b128, IOFFSET applies to both addresses)
__device__ __forceinline__ void due_async_copy32(unsigned lds_byte, const void* gptr) {
    unsigned long long ga = (unsigned long long)(uintptr_t)gptr;
    asm volatile("global_load_async_to_lds_b128 %0, %1, off\n\t"
                 "global_load_async_to_lds_b128 %0, %1, off offset:16"
                 :: "v"(lds_byte), "v"(ga) : "memory");
}
__device__ __forceinline__ void due_wait_async() {
    asm volatile("s_wait_asynccnt 0x0" ::: "memory");
}

// ---------------- power iteration helpers ----------------
__global__ void due_set_ones(float* x, int n) {
    int i = blockIdx.x * blockDim.x + threadIdx.x;
    if (i < n) x[i] = 1.0f;
}

__global__ __launch_bounds__(256) void due_matvec_col(const float* __restrict__ W,
                                                      const float* __restrict__ u,
                                                      float* __restrict__ v,
                                                      int rows, int cols) {
    int c = blockIdx.x * blockDim.x + threadIdx.x;
    if (c >= cols) return;
    float s = 0.0f;
    for (int r = 0; r < rows; ++r) s += W[(size_t)r * cols + c] * u[r];
    v[c] = s;
}

__global__ __launch_bounds__(256) void due_matvec_row(const float* __restrict__ W,
                                                      const float* __restrict__ v,
                                                      float* __restrict__ u,
                                                      int rows, int cols) {
    __shared__ float red[256];
    int r = blockIdx.x;
    float s = 0.0f;
    for (int c = threadIdx.x; c < cols; c += 256) s += W[(size_t)r * cols + c] * v[c];
    red[threadIdx.x] = s; __syncthreads();
    for (int st = 128; st > 0; st >>= 1) {
        if (threadIdx.x < st) red[threadIdx.x] += red[threadIdx.x + st];
        __syncthreads();
    }
    if (threadIdx.x == 0) u[r] = red[0];
}

__global__ __launch_bounds__(256) void due_normalize(float* __restrict__ x, int n,
                                                     float* __restrict__ inv_out) {
    __shared__ float red[256];
    float s = 0.0f;
    for (int i = threadIdx.x; i < n; i += 256) { float t = x[i]; s += t * t; }
    red[threadIdx.x] = s; __syncthreads();
    for (int st = 128; st > 0; st >>= 1) {
        if (threadIdx.x < st) red[threadIdx.x] += red[threadIdx.x + st];
        __syncthreads();
    }
    float inv = 1.0f / (sqrtf(red[0]) + 1e-20f);
    for (int i = threadIdx.x; i < n; i += 256) x[i] *= inv;
    if (threadIdx.x == 0 && inv_out) *inv_out = inv;
}

// ---------------- LayerNorm rows -> bf16 ----------------
__global__ __launch_bounds__(256) void due_ln_rows(const float* __restrict__ x,
                                                   const float* __restrict__ g,
                                                   const float* __restrict__ b,
                                                   __bf16* __restrict__ y, int n) {
    __shared__ float red[256];
    __shared__ float s_mean, s_rstd;
    int tid = threadIdx.x;
    const float* row = x + (size_t)blockIdx.x * n;
    float s = 0.0f;
    for (int i = tid; i < n; i += 256) s += row[i];
    red[tid] = s; __syncthreads();
    for (int st = 128; st > 0; st >>= 1) { if (tid < st) red[tid] += red[tid + st]; __syncthreads(); }
    if (tid == 0) s_mean = red[0] / (float)n;
    __syncthreads();
    float m = s_mean;
    float v = 0.0f;
    for (int i = tid; i < n; i += 256) { float d = row[i] - m; v += d * d; }
    red[tid] = v; __syncthreads();
    for (int st = 128; st > 0; st >>= 1) { if (tid < st) red[tid] += red[tid + st]; __syncthreads(); }
    if (tid == 0) s_rstd = rsqrtf(red[0] / (float)n + 1e-5f);
    __syncthreads();
    float rs = s_rstd;
    __bf16* yr = y + (size_t)blockIdx.x * n;
    for (int i = tid; i < n; i += 256)
        yr[i] = (__bf16)((row[i] - m) * rs * g[i] + b[i]);
}

// ---------------- transpose + scale -> bf16 : out[c][r] = W[r][c]*sc ----------------
__global__ __launch_bounds__(256) void due_transpose_bf16(const float* __restrict__ W,
                                                          __bf16* __restrict__ out,
                                                          int rows, int cols,
                                                          const float* __restrict__ invs) {
    __shared__ float tile[32][33];
    int tx = threadIdx.x & 31, ty = threadIdx.x >> 5;
    float sc = invs ? *invs : 1.0f;
    int c = blockIdx.x * 32 + tx;
    for (int rr = ty; rr < 32; rr += 8) {
        int r = blockIdx.y * 32 + rr;
        tile[rr][tx] = (r < rows && c < cols) ? W[(size_t)r * cols + c] : 0.0f;
    }
    __syncthreads();
    int r2 = blockIdx.y * 32 + tx;
    for (int cc = ty; cc < 32; cc += 8) {
        int c2 = blockIdx.x * 32 + cc;
        if (c2 < cols && r2 < rows)
            out[(size_t)c2 * rows + r2] = (__bf16)(tile[tx][cc] * sc);
    }
}

// ---------------- WMMA bf16 GEMM, 128x128 block tile, double-buffered async ----------------
// epi: 0 = +bias ; 1 = +bias,GELU ; 3 = RBF kernel epilogue -> f32
__global__ __launch_bounds__(256) void due_gemm_bf16(
    const __bf16* __restrict__ A,  // [M, Kd] row-major bf16
    const __bf16* __restrict__ Bw, // [Kd, N] row-major bf16
    const float* __restrict__ bias,
    float* __restrict__ Cf, __bf16* __restrict__ Cb,
    int M, int N, int Kd, int epi,
    const float* __restrict__ h2, const float* __restrict__ z2,
    const float* __restrict__ lls) {
    __shared__ __bf16 As[2][128][40];   // row-major A tiles (async-filled)
    __shared__ __bf16 Bs[2][128][40];   // COLUMN-major B tiles: Bs[p][n][k]

    const int tid  = threadIdx.x;
    const int lane = tid & 31;
    const int wid  = tid >> 5;
    const int wrow = wid >> 1;                 // 0..3 -> 32 rows each
    const int wcol = wid & 1;                  // 0..1 -> 64 cols each
    const int m_blk = blockIdx.y * 128;
    const int n_blk = blockIdx.x * 128;

    f32x8 acc[2][4];
#pragma unroll
    for (int i = 0; i < 2; ++i)
#pragma unroll
        for (int j = 0; j < 4; ++j)
#pragma unroll
            for (int r = 0; r < 8; ++r) acc[i][j][r] = 0.0f;

    const int ar = tid >> 1;                   // A tile row 0..127
    const int ac = (tid & 1) * 16;             // A k-offset 0/16
    const int br = tid >> 3;                   // B tile k-row 0..31
    const int bc = (tid & 7) * 16;             // B col offset 0..112
    const int gn = n_blk + bc;                 // guarded B column (N % 16 == 0)

    const int nk = Kd >> 5;

    // ---- prologue: stage tile 0 into buffer 0 ----
    {
        bf16x8 q0, q1;
        if (gn < N) {
            const __bf16* src = Bw + (size_t)br * N + gn;
            q0 = *(const bf16x8*)src;
            q1 = *(const bf16x8*)(src + 8);
        } else {
#pragma unroll
            for (int i = 0; i < 8; ++i) { q0[i] = (__bf16)0.0f; q1[i] = (__bf16)0.0f; }
        }
        due_async_copy32((unsigned)(uintptr_t)&As[0][ar][ac],
                         A + (size_t)(m_blk + ar) * Kd + ac);
#pragma unroll
        for (int i = 0; i < 8; ++i) {
            Bs[0][bc + i][br]     = q0[i];
            Bs[0][bc + 8 + i][br] = q1[i];
        }
        due_wait_async();
        __syncthreads();
    }

    for (int t = 0; t < nk; ++t) {
        const int p = t & 1;

        // fragments from buffer p
        bf16x16 af[2];
#pragma unroll
        for (int i = 0; i < 2; ++i) {
            int m  = wrow * 32 + i * 16 + (lane & 15);
            int kb = (lane < 16) ? 0 : 8;
            bf16x8 lo = *(const bf16x8*)&As[p][m][kb];
            bf16x8 hi = *(const bf16x8*)&As[p][m][16 + kb];
#pragma unroll
            for (int e = 0; e < 8; ++e) { af[i][e] = lo[e]; af[i][8 + e] = hi[e]; }
        }
        bf16x16 bfg[4];
#pragma unroll
        for (int j = 0; j < 4; ++j) {
            int n  = wcol * 64 + j * 16 + (lane & 15);
            int kb = (lane < 16) ? 0 : 8;
            bf16x8 lo = *(const bf16x8*)&Bs[p][n][kb];
            bf16x8 hi = *(const bf16x8*)&Bs[p][n][16 + kb];
#pragma unroll
            for (int e = 0; e < 8; ++e) { bfg[j][e] = lo[e]; bfg[j][8 + e] = hi[e]; }
        }

        // issue next-tile loads (global B -> VGPRs, async A -> LDS) before the WMMAs
        const bool more = (t + 1 < nk);
        bf16x8 q0, q1;
        if (more) {
            int k0n = (t + 1) << 5;
            if (gn < N) {
                const __bf16* src = Bw + (size_t)(k0n + br) * N + gn;
                q0 = *(const bf16x8*)src;
                q1 = *(const bf16x8*)(src + 8);
            } else {
#pragma unroll
                for (int i = 0; i < 8; ++i) { q0[i] = (__bf16)0.0f; q1[i] = (__bf16)0.0f; }
            }
            due_async_copy32((unsigned)(uintptr_t)&As[1 - p][ar][ac],
                             A + (size_t)(m_blk + ar) * Kd + k0n + ac);
        }

#pragma unroll
        for (int i = 0; i < 2; ++i)
#pragma unroll
            for (int j = 0; j < 4; ++j)
                acc[i][j] = __builtin_amdgcn_wmma_f32_16x16x32_bf16(
                    false, af[i], false, bfg[j], (short)0, acc[i][j], false, false);

        if (more) {
#pragma unroll
            for (int i = 0; i < 8; ++i) {
                Bs[1 - p][bc + i][br]     = q0[i];
                Bs[1 - p][bc + 8 + i][br] = q1[i];
            }
            due_wait_async();
        }
        __syncthreads();
    }

    float ls2inv = 0.0f;
    if (epi == 3) ls2inv = 1.0f / (2.0f * __expf(2.0f * (*lls)));

#pragma unroll
    for (int i = 0; i < 2; ++i) {
        int mbase = m_blk + wrow * 32 + i * 16 + ((lane < 16) ? 0 : 8);
#pragma unroll
        for (int j = 0; j < 4; ++j) {
            int col = n_blk + wcol * 64 + j * 16 + (lane & 15);
            if (col >= N) continue;
            float bv = bias ? bias[col] : 0.0f;
#pragma unroll
            for (int r = 0; r < 8; ++r) {
                int row = mbase + r;
                float a = acc[i][j][r];
                float v = a + bv;
                if (epi == 1) v = gelu_exact(v);
                else if (epi == 3) {
                    float d2 = fmaxf(h2[row] + z2[col] - 2.0f * a, 0.0f);
                    v = __expf(-d2 * ls2inv);
                }
                if (Cf) Cf[(size_t)row * N + col] = v;
                if (Cb) Cb[(size_t)row * N + col] = (__bf16)v;
            }
        }
    }
}

// ---------------- per-class kzz / Cholesky / alpha / z2 ----------------
__global__ __launch_bounds__(64) void due_chol_alpha(const float* __restrict__ z,
                                                     const float* __restrict__ vm,
                                                     const float* __restrict__ lls,
                                                     float* __restrict__ Lout,
                                                     float* __restrict__ alpha,
                                                     float* __restrict__ z2out) {
    __shared__ float kzz[20][20];
    __shared__ float zsh[20][128];
    __shared__ float z2s[20];
    int c = blockIdx.x, tid = threadIdx.x;
    const float* zc = z + (size_t)c * 20 * 128;
    for (int i = tid; i < 20 * 128; i += 64) zsh[i / 128][i % 128] = zc[i];
    __syncthreads();
    if (tid < 20) {
        float s = 0.0f;
        for (int k = 0; k < 128; ++k) { float t = zsh[tid][k]; s += t * t; }
        z2s[tid] = s;
        z2out[c * 20 + tid] = s;
    }
    __syncthreads();
    float ls2 = __expf(2.0f * (*lls));
    for (int p = tid; p < 400; p += 64) {
        int n = p / 20, m = p % 20;
        float s = 0.0f;
        for (int k = 0; k < 128; ++k) s += zsh[n][k] * zsh[m][k];
        float d2 = fmaxf(z2s[n] + z2s[m] - 2.0f * s, 0.0f);
        float kv = __expf(-d2 / (2.0f * ls2));
        if (n == m) kv += 1e-4f;
        kzz[n][m] = kv;
    }
    __syncthreads();
    if (tid == 0) {
        for (int i = 0; i < 20; ++i) {
            for (int j = 0; j <= i; ++j) {
                float s = kzz[i][j];
                for (int k = 0; k < j; ++k) s -= kzz[i][k] * kzz[j][k];
                if (i == j) kzz[i][i] = sqrtf(fmaxf(s, 1e-10f));
                else        kzz[i][j] = s / kzz[j][j];
            }
        }
        float y[20], a[20];
        for (int i = 0; i < 20; ++i) {
            float s = vm[c * 20 + i];
            for (int k = 0; k < i; ++k) s -= kzz[i][k] * y[k];
            y[i] = s / kzz[i][i];
        }
        for (int i = 19; i >= 0; --i) {
            float s = y[i];
            for (int k = i + 1; k < 20; ++k) s -= kzz[k][i] * a[k];
            a[i] = s / kzz[i][i];
        }
        for (int i = 0; i < 20; ++i) {
            alpha[c * 20 + i] = a[i];
            for (int j = 0; j < 20; ++j)
                Lout[(size_t)c * 400 + i * 20 + j] = (j <= i) ? kzz[i][j] : 0.0f;
        }
    }
}

// ---------------- h2[b] = ||h_b||^2 ----------------
__global__ __launch_bounds__(128) void due_h2(const float* __restrict__ h, float* __restrict__ h2) {
    __shared__ float red[128];
    int b = blockIdx.x;
    float t = h[(size_t)b * 128 + threadIdx.x];
    red[threadIdx.x] = t * t; __syncthreads();
    for (int st = 64; st > 0; st >>= 1) {
        if (threadIdx.x < st) red[threadIdx.x] += red[threadIdx.x + st];
        __syncthreads();
    }
    if (threadIdx.x == 0) h2[b] = red[0];
}

// ---------------- GP finish: logits, softmax, uncertainty ----------------
__global__ __launch_bounds__(256) void due_gp_final(const float* __restrict__ kxz, // [B,4000]
                                                    const float* __restrict__ L,   // [200,20,20]
                                                    const float* __restrict__ alpha,
                                                    float* __restrict__ probs,
                                                    float* __restrict__ logits,
                                                    float* __restrict__ unc) {
    int b = blockIdx.x * 256 + threadIdx.x;
    if (b >= 4096) return;
    const float* kr = kxz + (size_t)b * 4000;
    float mx = -1e30f, usum = 0.0f;
    for (int c = 0; c < 200; ++c) {
        const float* kv = kr + c * 20;
        const float* Lc = L + (size_t)c * 400;
        const float* ac = alpha + c * 20;
        float y[20];
        float lg = 0.0f, vs = 0.0f;
#pragma unroll
        for (int i = 0; i < 20; ++i) {
            float ki = kv[i];
            lg += ki * ac[i];
            float s = ki;
#pragma unroll
            for (int k = 0; k < i; ++k) s -= Lc[i * 20 + k] * y[k];
            y[i] = s / Lc[i * 20 + i];
            vs += y[i] * y[i];
        }
        usum += fmaxf(1.0f - vs, 0.0f);
        logits[(size_t)b * 200 + c] = lg;
        mx = fmaxf(mx, lg);
    }
    unc[b] = usum * (1.0f / 200.0f);
    float se = 0.0f;
    for (int c = 0; c < 200; ++c) {
        float e = __expf(logits[(size_t)b * 200 + c] - mx);
        probs[(size_t)b * 200 + c] = e;
        se += e;
    }
    float inv = 1.0f / se;
    for (int c = 0; c < 200; ++c) probs[(size_t)b * 200 + c] *= inv;
}

// ---------------- host launch ----------------
extern "C" void kernel_launch(void* const* d_in, const int* in_sizes, int n_in,
                              void* d_out, int out_size, void* d_ws, size_t ws_size,
                              hipStream_t stream) {
    (void)in_sizes; (void)n_in; (void)out_size; (void)ws_size;
    const int Bn = 4096, E = 768, Hd = 2048, Kd = 128;

    const float* features = (const float*)d_in[0];
    const float* ln1_g    = (const float*)d_in[1];
    const float* ln1_b    = (const float*)d_in[2];
    const float* W1       = (const float*)d_in[3];
    const float* b1       = (const float*)d_in[4];
    const float* ln2_g    = (const float*)d_in[5];
    const float* ln2_b    = (const float*)d_in[6];
    const float* W2       = (const float*)d_in[7];
    const float* b2       = (const float*)d_in[8];
    const float* Wp       = (const float*)d_in[9];
    const float* bp       = (const float*)d_in[10];
    const float* lls      = (const float*)d_in[11];
    const float* zind     = (const float*)d_in[12];
    const float* vmean    = (const float*)d_in[13];

    char* ws = (char*)d_ws;
    size_t off = 0;
    auto take = [&](size_t bytes) -> char* {
        char* p = ws + off;
        off += (bytes + 255) & ~(size_t)255;
        return p;
    };
    float*  invsig = (float*)take(16 * 4);
    float*  pu     = (float*)take(2048 * 4);
    float*  pv     = (float*)take(2048 * 4);
    char*   reuse  = take(64ull << 20);                 // 64MB aliased region
    float*  H1     = (float*)reuse;                     // 32MB: gemm1 f32 out
    __bf16* ABUF   = (__bf16*)(reuse + (32ull << 20));  // 16MB: LN outputs (bf16)
    __bf16* A2BUF  = (__bf16*)(reuse + (48ull << 20));  // 16MB: gemm2 bf16 out
    float*  KXZ    = (float*)reuse;                     // 62.5MB alias (h1/ABUF/A2BUF dead)
    __bf16* WT1    = (__bf16*)take((size_t)E  * Hd * 2);
    __bf16* WT2    = (__bf16*)take((size_t)Hd * Hd * 2);
    __bf16* WTP    = (__bf16*)take((size_t)Hd * Kd * 2);
    __bf16* ZT     = (__bf16*)take((size_t)Kd * 4000 * 2);
    float*  HBUF   = (float*)take((size_t)Bn * Kd * 4);
    __bf16* HBF    = (__bf16*)take((size_t)Bn * Kd * 2);
    float*  H2v    = (float*)take((size_t)Bn * 4);
    float*  Z2v    = (float*)take(4000 * 4);
    float*  LMAT   = (float*)take((size_t)200 * 400 * 4);
    float*  ALPHA  = (float*)take((size_t)200 * 20 * 4);

    float* probs  = (float*)d_out;
    float* logits = probs + (size_t)Bn * 200;
    float* unc    = probs + (size_t)2 * Bn * 200;

    // 1) spectral norms (power iteration, 8 iters each)
    struct { const float* W; int rows, cols, slot; } mats[3] = {
        {W1, Hd, E, 0}, {W2, Hd, Hd, 1}, {Wp, Kd, Hd, 2}};
    for (int j = 0; j < 3; ++j) {
        int rows = mats[j].rows, cols = mats[j].cols;
        due_set_ones<<<(rows + 255) / 256, 256, 0, stream>>>(pu, rows);
        for (int it = 0; it < 8; ++it) {
            due_matvec_col<<<(cols + 255) / 256, 256, 0, stream>>>(mats[j].W, pu, pv, rows, cols);
            due_normalize<<<1, 256, 0, stream>>>(pv, cols, nullptr);
            due_matvec_row<<<rows, 256, 0, stream>>>(mats[j].W, pv, pu, rows, cols);
            due_normalize<<<1, 256, 0, stream>>>(pu, rows, (it == 7) ? (invsig + mats[j].slot) : nullptr);
        }
    }

    // 2) weight / inducing-point conversion to bf16 B-operands
    due_transpose_bf16<<<dim3((E + 31) / 32, (Hd + 31) / 32), 256, 0, stream>>>(W1, WT1, Hd, E, invsig + 0);
    due_transpose_bf16<<<dim3((Hd + 31) / 32, (Hd + 31) / 32), 256, 0, stream>>>(W2, WT2, Hd, Hd, invsig + 1);
    due_transpose_bf16<<<dim3((Hd + 31) / 32, (Kd + 31) / 32), 256, 0, stream>>>(Wp, WTP, Kd, Hd, invsig + 2);
    due_transpose_bf16<<<dim3((Kd + 31) / 32, (4000 + 31) / 32), 256, 0, stream>>>(zind, ZT, 4000, Kd, nullptr);

    // 3) MLP (WMMA bf16 GEMMs, fused bias/GELU epilogues)
    due_ln_rows<<<Bn, 256, 0, stream>>>(features, ln1_g, ln1_b, ABUF, E);
    due_gemm_bf16<<<dim3(Hd / 128, Bn / 128), 256, 0, stream>>>(
        ABUF, WT1, b1, H1, nullptr, Bn, Hd, E, /*epi=*/1, nullptr, nullptr, nullptr);
    due_ln_rows<<<Bn, 256, 0, stream>>>(H1, ln2_g, ln2_b, ABUF, Hd);
    due_gemm_bf16<<<dim3(Hd / 128, Bn / 128), 256, 0, stream>>>(
        ABUF, WT2, b2, nullptr, A2BUF, Bn, Hd, Hd, /*epi=*/1, nullptr, nullptr, nullptr);
    due_gemm_bf16<<<dim3(Kd / 128, Bn / 128), 256, 0, stream>>>(
        A2BUF, WTP, bp, HBUF, HBF, Bn, Kd, Hd, /*epi=*/0, nullptr, nullptr, nullptr);

    // 4) GP pre-computation
    due_chol_alpha<<<200, 64, 0, stream>>>(zind, vmean, lls, LMAT, ALPHA, Z2v);
    due_h2<<<Bn, 128, 0, stream>>>(HBUF, H2v);

    // 5) k_xz = exp(-d2/2ls^2) via WMMA GEMM with fused RBF epilogue  [Bn, 4000]
    due_gemm_bf16<<<dim3((4000 + 127) / 128, Bn / 128), 256, 0, stream>>>(
        HBF, ZT, nullptr, KXZ, nullptr, Bn, 4000, Kd, /*epi=*/3, H2v, Z2v, lls);

    // 6) logits / probs / uncertainty
    due_gp_final<<<Bn / 256, 256, 0, stream>>>(KXZ, LMAT, ALPHA, probs, logits, unc);
}
